// TransformerBlock_26456998543686
// MI455X (gfx1250) — matrix operand order repaired and checked by reading, not compiled, not run on hardware
//
#include <hip/hip_runtime.h>
#include <hip/hip_bf16.h>
#include <stdint.h>

// ---------------------------------------------------------------------------
// MI455X (gfx1250) transformer block.
// All GEMM-shaped work (DFT-as-GEMM, QKV, attention, output proj, GLU FFN)
// runs through v_wmma_f32_16x16x32_bf16 (f32 accumulate). wave32 everywhere.
// GEMMs use double-buffered LDS (one barrier per K-step, global loads
// overlapped with WMMA). Epilogues use v_rcp_f32 instead of IEEE division.
// ---------------------------------------------------------------------------

typedef __attribute__((ext_vector_type(16))) __bf16 v16bf;
typedef __attribute__((ext_vector_type(8)))  __bf16 v8bf;
typedef __attribute__((ext_vector_type(8)))  float  v8f;

union V16 { v16bf v; v8bf h[2]; };

__device__ __forceinline__ unsigned short f2bf(float f) {
  unsigned int u = __float_as_uint(f);
  unsigned int r = u + 0x7FFFu + ((u >> 16) & 1u);
  return (unsigned short)(r >> 16);
}

__device__ __forceinline__ float fast_rcp(float x) {
  return __builtin_amdgcn_rcpf(x);          // v_rcp_f32
}

__device__ __forceinline__ v8f wmma_bf16(const V16& a, const V16& b, v8f c) {
  return __builtin_amdgcn_wmma_f32_16x16x32_bf16(
      /*neg_a=*/false, a.v, /*neg_b=*/false, b.v,
      /*c_mod=*/(short)0, c, /*reuse_a=*/false, /*reuse_b=*/false);
}

enum : int { EB16 = 1, EBIAS = 2, ERES = 4, ENEG = 8 };

// ---------------------------------------------------------------------------
// Generic bf16 GEMM: Out[M x N] = op(A[M x K] @ B[K x N]) (+bias)(+res)
// Block tile 128x64, BK=32, 8 waves (4x2), wave tile 32x32 (2x2 WMMA frags).
// Double-buffered LDS: one s_barrier per K-step.
// BATCHED: A is shared (DFT matrix, M=1024 local rows), B/Out offset by
// blockIdx.z*1024 rows (per-batch GEMM along the sequence axis).
// ---------------------------------------------------------------------------
template<int EPI, bool BATCHED>
__global__ __launch_bounds__(256)
void gemm_bf16_kernel(const unsigned short* __restrict__ A, int lda,
                      const unsigned short* __restrict__ B, int ldb,
                      void* __restrict__ Out, int ldo,
                      const float* __restrict__ bias,
                      const float* __restrict__ res, int ldr,
                      int K)
{
  constexpr int BM = 128, BN = 64, BK = 32;
  __shared__ unsigned short As[2][BM][BK];   // row-major (M x K)
  __shared__ unsigned short Bs[2][BN][BK];   // transposed: col n contiguous in K

  const int tid  = threadIdx.x;
  const int lane = tid & 31;
  const int wave = tid >> 5;
  const int wm   = wave >> 1;          // 0..3
  const int wn   = wave & 1;           // 0..1
  const int lr   = lane & 15;
  const int hi   = lane >> 4;          // 0 or 1

  const int m0 = blockIdx.y * BM;
  const int n0 = blockIdx.x * BN;
  const int zR = BATCHED ? (int)blockIdx.z * 1024 : 0;  // B / Out row offset

  const int arow = tid >> 1;            // 0..127
  const int akp  = (tid & 1) * 16;      // 0 or 16
  const int bkk  = tid >> 3;            // 0..31
  const int bnn  = (tid & 7) * 8;       // 0..56

  const unsigned short* gA = A + (size_t)(m0 + arow) * lda + akp;
  const unsigned short* gB = B + (size_t)(zR + bkk) * ldb + (n0 + bnn);

  v8f acc[2][2] = {};

  // ---- prologue: stage tile 0 ----
  uint4 ra0 = *(const uint4*)gA;
  uint4 ra1 = *(const uint4*)(gA + 8);
  uint4 rb  = *(const uint4*)gB;
  {
    *(uint4*)&As[0][arow][akp]     = ra0;
    *(uint4*)&As[0][arow][akp + 8] = ra1;
    unsigned short tmp[8]; *(uint4*)tmp = rb;
    #pragma unroll
    for (int j = 0; j < 8; ++j) Bs[0][bnn + j][bkk] = tmp[j];
  }
  __syncthreads();

  int cur = 0;
  for (int k0 = 0; k0 < K; k0 += BK) {
    const bool hasNext = (k0 + BK) < K;
    // ---- issue next tile's global loads early (hidden under WMMA) ----
    if (hasNext) {
      ra0 = *(const uint4*)(gA + (k0 + BK));
      ra1 = *(const uint4*)(gA + (k0 + BK) + 8);
      rb  = *(const uint4*)(gB + (size_t)(k0 + BK) * ldb);
    }

    // ---- WMMA pack on current buffer (CDNA5 16-bit A/B layouts) ----
    #pragma unroll
    for (int s = 0; s < 2; ++s) {
      V16 a;
      const int row = wm * 32 + s * 16 + lr;
      a.h[0] = *(const v8bf*)&As[cur][row][hi * 8];        // K 0..7  / 8..15
      a.h[1] = *(const v8bf*)&As[cur][row][16 + hi * 8];   // K 16..23/ 24..31
      #pragma unroll
      for (int t = 0; t < 2; ++t) {
        V16 b;
        const int col = wn * 32 + t * 16 + lr;
        b.h[0] = *(const v8bf*)&Bs[cur][col][hi * 8];      // K 0..7  / 8..15
        b.h[1] = *(const v8bf*)&Bs[cur][col][16 + hi * 8];
        acc[s][t] = wmma_bf16(a, b, acc[s][t]);
      }
    }

    // ---- commit next tile to the other buffer; single barrier ----
    if (hasNext) {
      const int nxt = cur ^ 1;
      *(uint4*)&As[nxt][arow][akp]     = ra0;
      *(uint4*)&As[nxt][arow][akp + 8] = ra1;
      unsigned short tmp[8]; *(uint4*)tmp = rb;
      #pragma unroll
      for (int j = 0; j < 8; ++j) Bs[nxt][bnn + j][bkk] = tmp[j];
      __syncthreads();
      cur = nxt;
    }
  }

  // ---- epilogue (coalesced per half-wave: lanes 0-15 write one row) ----
  #pragma unroll
  for (int s = 0; s < 2; ++s) {
    #pragma unroll
    for (int t = 0; t < 2; ++t) {
      const int col = n0 + wn * 32 + t * 16 + lr;
      #pragma unroll
      for (int r = 0; r < 8; ++r) {
        const int row = m0 + wm * 32 + s * 16 + r + hi * 8;
        float v = acc[s][t][r];
        if (EPI & ENEG)  v = -v;
        if (EPI & EBIAS) v += bias[col];
        if (EPI & ERES)  v += res[(size_t)(zR + row) * ldr + col];
        if (EPI & EB16)
          ((unsigned short*)Out)[(size_t)(zR + row) * ldo + col] = f2bf(v);
        else
          ((float*)Out)[(size_t)(zR + row) * ldo + col] = v;
      }
    }
  }
}

// ---------------------------------------------------------------------------
// GLU FFN GEMM: h = silu(Xn @ Wu) * (Xn @ Wv), Wu/Wv = w_uv[:, :FF]/[:, FF:].
// Same tiling (double buffered), dual accumulators; silu via v_rcp_f32.
// ---------------------------------------------------------------------------
__global__ __launch_bounds__(256)
void gemm_glu_kernel(const unsigned short* __restrict__ A, int lda,
                     const unsigned short* __restrict__ B, int ldb, int voff,
                     unsigned short* __restrict__ Out, int ldo, int K)
{
  constexpr int BM = 128, BN = 64, BK = 32;
  __shared__ unsigned short As[2][BM][BK];
  __shared__ unsigned short Bu[2][BN][BK];
  __shared__ unsigned short Bv[2][BN][BK];

  const int tid  = threadIdx.x;
  const int lane = tid & 31;
  const int wave = tid >> 5;
  const int wm   = wave >> 1, wn = wave & 1;
  const int lr   = lane & 15, hi = lane >> 4;
  const int m0 = blockIdx.y * BM;
  const int n0 = blockIdx.x * BN;

  const int arow = tid >> 1, akp = (tid & 1) * 16;
  const int bkk  = tid >> 3, bnn = (tid & 7) * 8;

  const unsigned short* gA = A + (size_t)(m0 + arow) * lda + akp;
  const unsigned short* gB = B + (size_t)bkk * ldb + (n0 + bnn);

  v8f accu[2][2] = {};
  v8f accv[2][2] = {};

  uint4 ra0 = *(const uint4*)gA;
  uint4 ra1 = *(const uint4*)(gA + 8);
  uint4 rbu = *(const uint4*)gB;
  uint4 rbv = *(const uint4*)(gB + voff);
  {
    *(uint4*)&As[0][arow][akp]     = ra0;
    *(uint4*)&As[0][arow][akp + 8] = ra1;
    unsigned short tu[8], tv[8];
    *(uint4*)tu = rbu; *(uint4*)tv = rbv;
    #pragma unroll
    for (int j = 0; j < 8; ++j) { Bu[0][bnn + j][bkk] = tu[j]; Bv[0][bnn + j][bkk] = tv[j]; }
  }
  __syncthreads();

  int cur = 0;
  for (int k0 = 0; k0 < K; k0 += BK) {
    const bool hasNext = (k0 + BK) < K;
    if (hasNext) {
      ra0 = *(const uint4*)(gA + (k0 + BK));
      ra1 = *(const uint4*)(gA + (k0 + BK) + 8);
      rbu = *(const uint4*)(gB + (size_t)(k0 + BK) * ldb);
      rbv = *(const uint4*)(gB + (size_t)(k0 + BK) * ldb + voff);
    }

    #pragma unroll
    for (int s = 0; s < 2; ++s) {
      V16 a;
      const int row = wm * 32 + s * 16 + lr;
      a.h[0] = *(const v8bf*)&As[cur][row][hi * 8];
      a.h[1] = *(const v8bf*)&As[cur][row][16 + hi * 8];
      #pragma unroll
      for (int t = 0; t < 2; ++t) {
        const int col = wn * 32 + t * 16 + lr;
        V16 bu, bv;
        bu.h[0] = *(const v8bf*)&Bu[cur][col][hi * 8];
        bu.h[1] = *(const v8bf*)&Bu[cur][col][16 + hi * 8];
        bv.h[0] = *(const v8bf*)&Bv[cur][col][hi * 8];
        bv.h[1] = *(const v8bf*)&Bv[cur][col][16 + hi * 8];
        accu[s][t] = wmma_bf16(a, bu, accu[s][t]);
        accv[s][t] = wmma_bf16(a, bv, accv[s][t]);
      }
    }

    if (hasNext) {
      const int nxt = cur ^ 1;
      *(uint4*)&As[nxt][arow][akp]     = ra0;
      *(uint4*)&As[nxt][arow][akp + 8] = ra1;
      unsigned short tu[8], tv[8];
      *(uint4*)tu = rbu; *(uint4*)tv = rbv;
      #pragma unroll
      for (int j = 0; j < 8; ++j) { Bu[nxt][bnn + j][bkk] = tu[j]; Bv[nxt][bnn + j][bkk] = tv[j]; }
      __syncthreads();
      cur = nxt;
    }
  }

  #pragma unroll
  for (int s = 0; s < 2; ++s) {
    #pragma unroll
    for (int t = 0; t < 2; ++t) {
      const int col = n0 + wn * 32 + t * 16 + lr;
      #pragma unroll
      for (int r = 0; r < 8; ++r) {
        const int row = m0 + wm * 32 + s * 16 + r + hi * 8;
        const float u = accu[s][t][r];
        const float v = accv[s][t][r];
        const float h = u * fast_rcp(1.0f + __expf(-u)) * v;  // silu(u)*v
        Out[(size_t)row * ldo + col] = f2bf(h);
      }
    }
  }
}

// ---------------------------------------------------------------------------
// Flash attention: one wave32 per (b, h, 16-row q tile). Online softmax with
// 16-lane shfl_xor row reductions (rows live across lanes in the D layout).
// P is re-laid-out into an A fragment and V transposed via LDS.
// ---------------------------------------------------------------------------
__global__ __launch_bounds__(32)
void flash_attn_kernel(const unsigned short* __restrict__ Q,
                       const unsigned short* __restrict__ Kk,
                       const unsigned short* __restrict__ Vv,
                       unsigned short* __restrict__ O)
{
  constexpr int NSEQ = 1024, DMODEL = 1024;
  const int qt = blockIdx.x;      // 0..63  (16-row tile)
  const int h  = blockIdx.y;      // 0..15
  const int b  = blockIdx.z;      // 0..7
  const int lane = threadIdx.x;
  const int lr = lane & 15, hi = lane >> 4;

  const unsigned short* qb = Q  + (size_t)(b * NSEQ) * DMODEL + h * 64;
  const unsigned short* kb = Kk + (size_t)(b * NSEQ) * DMODEL + h * 64;
  const unsigned short* vb = Vv + (size_t)(b * NSEQ) * DMODEL + h * 64;

  // q fragments (A: 16 rows x K=64 split into two K=32 frags)
  V16 qf[2];
  {
    const unsigned short* qp = qb + (size_t)(qt * 16 + lr) * DMODEL;
    #pragma unroll
    for (int c = 0; c < 2; ++c) {
      qf[c].h[0] = *(const v8bf*)(qp + 32 * c + hi * 8);
      qf[c].h[1] = *(const v8bf*)(qp + 32 * c + 16 + hi * 8);
    }
  }

  float m_st[8], l_st[8];
  #pragma unroll
  for (int r = 0; r < 8; ++r) { m_st[r] = -1e30f; l_st[r] = 0.0f; }
  v8f o_acc[4] = {};

  __shared__ unsigned short Pls[16][32];  // P tile row-major for A-frag reload
  __shared__ unsigned short Vt[64][32];   // V transposed: [d][key]

  const float scale = 0.125f;             // 1/sqrt(64)

  for (int kt = 0; kt < NSEQ; kt += 32) {
    // ---- scores S = q @ k^T (16x32); k rows are K-contiguous == B columns
    v8f sc[2] = {};
    #pragma unroll
    for (int t = 0; t < 2; ++t) {
      #pragma unroll
      for (int c = 0; c < 2; ++c) {
        V16 bf;
        const unsigned short* kp =
            kb + (size_t)(kt + t * 16 + lr) * DMODEL + 32 * c + hi * 8;
        bf.h[0] = *(const v8bf*)kp;
        bf.h[1] = *(const v8bf*)(kp + 16);
        sc[t] = wmma_bf16(qf[c], bf, sc[t]);
      }
    }

    // ---- stage V tile transposed (32 keys x 64 d -> Vt[d][key])
    {
      const unsigned short* vp = vb + (size_t)(kt + lane) * DMODEL;
      #pragma unroll
      for (int d8 = 0; d8 < 8; ++d8) {
        uint4 dta = *(const uint4*)(vp + d8 * 8);
        unsigned short tmp[8]; *(uint4*)tmp = dta;
        #pragma unroll
        for (int j = 0; j < 8; ++j) Vt[d8 * 8 + j][lane] = tmp[j];
      }
    }

    // ---- online softmax (row r lives at lanes 0-15 (m=r) / 16-31 (m=r+8))
    float alpha[8];
    #pragma unroll
    for (int r = 0; r < 8; ++r) {
      float v = fmaxf(sc[0][r], sc[1][r]) * scale;
      #pragma unroll
      for (int w = 1; w < 16; w <<= 1) v = fmaxf(v, __shfl_xor(v, w, 16));
      const float mn = fmaxf(m_st[r], v);
      alpha[r] = __expf(m_st[r] - mn);
      m_st[r]  = mn;
    }
    #pragma unroll
    for (int t = 0; t < 2; ++t) {
      #pragma unroll
      for (int r = 0; r < 8; ++r) {
        const float p = __expf(sc[t][r] * scale - m_st[r]);
        Pls[r + hi * 8][t * 16 + lr] = f2bf(p);
        sc[t][r] = p;
      }
    }
    #pragma unroll
    for (int r = 0; r < 8; ++r) {
      float ps = sc[0][r] + sc[1][r];
      #pragma unroll
      for (int w = 1; w < 16; w <<= 1) ps += __shfl_xor(ps, w, 16);
      l_st[r] = l_st[r] * alpha[r] + ps;
      #pragma unroll
      for (int dt = 0; dt < 4; ++dt) o_acc[dt][r] *= alpha[r];
    }
    __syncthreads();

    // ---- O += P @ V  (A = P 16x32, B = Vt columns)
    V16 pf;
    pf.h[0] = *(const v8bf*)&Pls[lr][hi * 8];
    pf.h[1] = *(const v8bf*)&Pls[lr][16 + hi * 8];
    #pragma unroll
    for (int dt = 0; dt < 4; ++dt) {
      V16 bf;
      const int d = dt * 16 + lr;
      bf.h[0] = *(const v8bf*)&Vt[d][hi * 8];
      bf.h[1] = *(const v8bf*)&Vt[d][16 + hi * 8];
      o_acc[dt] = wmma_bf16(pf, bf, o_acc[dt]);
    }
    __syncthreads();
  }

  // ---- write o * (1/l)  (v_rcp_f32, hoisted per row)
  float inv_l[8];
  #pragma unroll
  for (int r = 0; r < 8; ++r) inv_l[r] = fast_rcp(l_st[r]);
  #pragma unroll
  for (int dt = 0; dt < 4; ++dt) {
    const int d = dt * 16 + lr;
    #pragma unroll
    for (int r = 0; r < 8; ++r) {
      const int row = qt * 16 + r + hi * 8;
      const float v = o_acc[dt][r] * inv_l[r];
      O[(size_t)(b * NSEQ + row) * DMODEL + h * 64 + d] = f2bf(v);
    }
  }
}

// ---------------------------------------------------------------------------
// Elementwise helpers
// ---------------------------------------------------------------------------
__global__ __launch_bounds__(256)
void cvt_f32_bf16_kernel(const float* __restrict__ in,
                         unsigned short* __restrict__ out, int n)
{
  int i = (blockIdx.x * 256 + threadIdx.x) * 4;
  if (i + 3 < n) {
    float4 v = *(const float4*)(in + i);
    out[i + 0] = f2bf(v.x); out[i + 1] = f2bf(v.y);
    out[i + 2] = f2bf(v.z); out[i + 3] = f2bf(v.w);
  }
}

__global__ __launch_bounds__(256)
void gen_dft_kernel(unsigned short* __restrict__ C, unsigned short* __restrict__ S)
{
  const int j = blockIdx.x;  // 0..1023
  for (int k = threadIdx.x; k < 1024; k += 256) {
    const int ph = (j * k) & 1023;
    const float ang = (float)ph * (6.2831853071795864769f / 1024.0f);
    C[j * 1024 + k] = f2bf(__cosf(ang));
    S[j * 1024 + k] = f2bf(__sinf(ang));
  }
}

__global__ __launch_bounds__(256)
void rmsnorm_kernel(const float* __restrict__ X, const float* __restrict__ scale,
                    unsigned short* __restrict__ Y)
{
  const int row = blockIdx.x;
  const float* x = X + (size_t)row * 1024;
  float vals[4];
  float s = 0.0f;
  #pragma unroll
  for (int i = 0; i < 4; ++i) {
    vals[i] = x[threadIdx.x + 256 * i];
    s += vals[i] * vals[i];
  }
  #pragma unroll
  for (int w = 1; w < 32; w <<= 1) s += __shfl_xor(s, w, 32);
  __shared__ float red[8];
  if ((threadIdx.x & 31) == 0) red[threadIdx.x >> 5] = s;
  __syncthreads();
  float tot = 0.0f;
  #pragma unroll
  for (int i = 0; i < 8; ++i) tot += red[i];
  const float r = rsqrtf(tot * (1.0f / 1024.0f) + 1e-5f);
  #pragma unroll
  for (int i = 0; i < 4; ++i) {
    const int c = threadIdx.x + 256 * i;
    Y[(size_t)row * 1024 + c] = f2bf(vals[i] * r * scale[c]);
  }
}

// ---------------------------------------------------------------------------
// Host orchestration
// ---------------------------------------------------------------------------
extern "C" void kernel_launch(void* const* d_in, const int* in_sizes, int n_in,
                              void* d_out, int out_size, void* d_ws, size_t ws_size,
                              hipStream_t stream)
{
  (void)in_sizes; (void)n_in; (void)out_size; (void)ws_size;
  const int D = 1024, NB = 8, NSEQ = 1024, BT = NB * NSEQ /*8192*/, FF = 2048;

  const float* x        = (const float*)d_in[0];
  const float* a_scale  = (const float*)d_in[1];
  const float* f_scale  = (const float*)d_in[2];
  const float* wq = (const float*)d_in[3];  const float* bq = (const float*)d_in[4];
  const float* wk = (const float*)d_in[5];  const float* bk = (const float*)d_in[6];
  const float* wv = (const float*)d_in[7];  const float* bv = (const float*)d_in[8];
  const float* wo = (const float*)d_in[9];  const float* bo = (const float*)d_in[10];
  const float* w_uv  = (const float*)d_in[11];
  const float* w_out = (const float*)d_in[12];

  // workspace layout (static offsets, 256B aligned; ~232 MB total)
  char* ws = (char*)d_ws;
  size_t off = 0;
  auto alloc = [&](size_t bytes) { size_t r = off; off += (bytes + 255) & ~(size_t)255; return r; };
  const size_t MB2 = (size_t)D * D * 2;
  unsigned short* wqb   = (unsigned short*)(ws + alloc(MB2));
  unsigned short* wkb   = (unsigned short*)(ws + alloc(MB2));
  unsigned short* wvb   = (unsigned short*)(ws + alloc(MB2));
  unsigned short* wob   = (unsigned short*)(ws + alloc(MB2));
  unsigned short* wuvb  = (unsigned short*)(ws + alloc((size_t)D * 2 * FF * 2));
  unsigned short* woutb = (unsigned short*)(ws + alloc((size_t)FF * D * 2));
  unsigned short* Cb    = (unsigned short*)(ws + alloc(MB2));
  unsigned short* Sb    = (unsigned short*)(ws + alloc(MB2));
  unsigned short* xb    = (unsigned short*)(ws + alloc((size_t)BT * D * 2));
  unsigned short* Pb    = (unsigned short*)(ws + alloc((size_t)BT * D * 2));
  unsigned short* Qb    = (unsigned short*)(ws + alloc((size_t)BT * D * 2));
  float*          xfft  = (float*)        (ws + alloc((size_t)BT * D * 4));
  unsigned short* xn    = (unsigned short*)(ws + alloc((size_t)BT * D * 2));
  unsigned short* qbuf  = (unsigned short*)(ws + alloc((size_t)BT * D * 2));
  unsigned short* kbuf  = (unsigned short*)(ws + alloc((size_t)BT * D * 2));
  unsigned short* vbuf  = (unsigned short*)(ws + alloc((size_t)BT * D * 2));
  unsigned short* obuf  = (unsigned short*)(ws + alloc((size_t)BT * D * 2));
  unsigned short* xn2   = (unsigned short*)(ws + alloc((size_t)BT * D * 2));
  unsigned short* hbuf  = (unsigned short*)(ws + alloc((size_t)BT * FF * 2));
  float* out_f = (float*)d_out;            // also serves as the x2 residual

  dim3 blk(256);
  auto cvt = [&](const float* src, unsigned short* dst, int n) {
    cvt_f32_bf16_kernel<<<dim3(n / 1024), blk, 0, stream>>>(src, dst, n);
  };
  cvt(wq, wqb, D * D);  cvt(wk, wkb, D * D);
  cvt(wv, wvb, D * D);  cvt(wo, wob, D * D);
  cvt(w_uv, wuvb, D * 2 * FF);
  cvt(w_out, woutb, FF * D);
  cvt(x, xb, BT * D);
  gen_dft_kernel<<<dim3(1024), blk, 0, stream>>>(Cb, Sb);

  const dim3 g_full(D / 64, BT / 128, 1);          // 8192 x 1024 GEMMs
  const dim3 g_batch(D / 64, NSEQ / 128, NB);      // per-batch 1024 x 1024 GEMMs

  // real(FFT2(x)) = C @ (x@C) - S @ (x@S)
  gemm_bf16_kernel<EB16, false><<<g_full, blk, 0, stream>>>(
      xb, D, Cb, D, Pb, D, nullptr, nullptr, 0, D);
  gemm_bf16_kernel<EB16, false><<<g_full, blk, 0, stream>>>(
      xb, D, Sb, D, Qb, D, nullptr, nullptr, 0, D);
  gemm_bf16_kernel<0, true><<<g_batch, blk, 0, stream>>>(
      Cb, D, Pb, D, xfft, D, nullptr, nullptr, 0, D);
  gemm_bf16_kernel<ENEG | ERES, true><<<g_batch, blk, 0, stream>>>(
      Sb, D, Qb, D, xfft, D, nullptr, xfft, D, D);

  // attention branch
  rmsnorm_kernel<<<dim3(BT), blk, 0, stream>>>(xfft, a_scale, xn);
  gemm_bf16_kernel<EB16 | EBIAS, false><<<g_full, blk, 0, stream>>>(
      xn, D, wqb, D, qbuf, D, bq, nullptr, 0, D);
  gemm_bf16_kernel<EB16 | EBIAS, false><<<g_full, blk, 0, stream>>>(
      xn, D, wkb, D, kbuf, D, bk, nullptr, 0, D);
  gemm_bf16_kernel<EB16 | EBIAS, false><<<g_full, blk, 0, stream>>>(
      xn, D, wvb, D, vbuf, D, bv, nullptr, 0, D);
  flash_attn_kernel<<<dim3(NSEQ / 16, 16, NB), dim3(32), 0, stream>>>(
      qbuf, kbuf, vbuf, obuf);
  // x2 = o @ wo + bo + xfft  (written to d_out, used as residual later)
  gemm_bf16_kernel<EBIAS | ERES, false><<<g_full, blk, 0, stream>>>(
      obuf, D, wob, D, out_f, D, bo, xfft, D, D);

  // FFN branch
  rmsnorm_kernel<<<dim3(BT), blk, 0, stream>>>(out_f, f_scale, xn2);
  gemm_glu_kernel<<<dim3(FF / 64, BT / 128, 1), blk, 0, stream>>>(
      xn2, D, wuvb, 2 * FF, FF, hbuf, FF, D);
  // out = h @ w_out + x2   (in-place residual read/write on d_out)
  gemm_bf16_kernel<ERES, false><<<g_full, blk, 0, stream>>>(
      hbuf, FF, woutb, D, out_f, D, nullptr, out_f, D, FF);
}
